// LSTM_36996848287879
// MI455X (gfx1250) — matrix operand order repaired and checked by reading, not compiled, not run on hardware
//
#include <hip/hip_runtime.h>
#include <hip/hip_bf16.h>
#include <math.h>

// ---------------------------------------------------------------------------
// LSTM trajectory predictor for MI455X (gfx1250), wave32 + WMMA f16->f32.
//
//  * wprep_kernel:     f32 [K,512] weights -> f16 CDNA5 B-operand fragments
//  * wprep_out_kernel: f32 [128,5] head weights -> f16 B-fragments (N pad 16)
//  * lstm_kernel: persistent per-workgroup tile of 64 tracks; c in VGPRs,
//    h (f32 + f16 A-fragments) in LDS; 19 timesteps looped in-kernel.
//      gates = embA @ W_ih + hA @ W_hh  (v_wmma_f32_16x16x32_f16)
//      raw   = hA @ W_out               (v_wmma_f32_16x16x32_f16, waves 0-3)
//    B fragments are loaded once per K-block and reused across all 4 M-tiles
//    (16 live accumulators) to minimize L2 traffic.
// ---------------------------------------------------------------------------

typedef __attribute__((ext_vector_type(16))) _Float16 v16h;
typedef __attribute__((ext_vector_type(8)))  float    v8f;

#define EMBD       64
#define HIDD       128
#define NTRACK     20000
#define ENC_STEPS  8      // OBS_LEN - 1
#define DEC_STEPS  11     // N_PREDICT - 1
#define TOT_STEPS  19
#define TM         64     // tracks per workgroup (4 M-tiles of 16)
#define BLK        256    // 8 wave32s

__device__ __forceinline__ float sigm_fast(float x) {
  return __fdividef(1.0f, 1.0f + __expf(-x));
}
__device__ __forceinline__ float tanh_fast(float x) {
  return fmaf(2.0f, sigm_fast(2.0f * x), -1.0f);
}
__device__ __forceinline__ float softplus_fast(float x) {
  return (x > 20.0f) ? x : __logf(1.0f + __expf(x));
}

// Convert a [K,512] f32 row-major weight matrix into f16 B-operand fragments.
// dst layout (halfs): ((kb*32 + nt)*32 + lane)*16 + q
// element(lane L, half q): N = nt*16 + (L%16), K = kb*32 + (L/16)*16 + q
__global__ void wprep_kernel(const float* __restrict__ src,
                             _Float16* __restrict__ dst, int K) {
  int total = K * 512;
  for (int idx = blockIdx.x * blockDim.x + threadIdx.x; idx < total;
       idx += gridDim.x * blockDim.x) {
    int q    = idx & 15;
    int ln   = (idx >> 4) & 31;
    int tile = idx >> 9;
    int nt   = tile & 31;
    int kb   = tile >> 5;
    int kk   = kb * 32 + ((ln >> 4) << 4) + q;   // K index
    int nn   = (nt << 4) + (ln & 15);            // N index
    dst[idx] = (_Float16)src[kk * 512 + nn];
  }
}

// W_out [128,5] -> f16 B fragments, single N-tile padded to 16 columns.
__global__ void wprep_out_kernel(const float* __restrict__ src,
                                 _Float16* __restrict__ dst) {
  int idx = blockIdx.x * blockDim.x + threadIdx.x;   // total 4*32*16 = 2048
  if (idx < 4 * 32 * 16) {
    int q  = idx & 15;
    int ln = (idx >> 4) & 31;
    int kb = idx >> 9;
    int kk = kb * 32 + ((ln >> 4) << 4) + q;
    int nn = ln & 15;
    dst[idx] = (_Float16)((nn < 5) ? src[kk * 5 + nn] : 0.0f);
  }
}

// A-operand fragment address: element (row m in 0..15 of M-tile, k)
//   k' = k%32 ; lane = m | (bit4(k')<<4) ; q = k'&15
__device__ __forceinline__ int a_frag_addr(int mt, int kbdim, int m, int k) {
  int kb = k >> 5, kp = k & 31;
  int ln = (m & 15) | ((kp >> 4) << 4);
  int q  = kp & 15;
  return (((mt * kbdim + kb) * 32 + ln) << 4) + q;
}

__global__ __launch_bounds__(BLK)
void lstm_kernel(const float* __restrict__ observed,   // [9, NT, 2]
                 const float* __restrict__ W_emb,      // [2, 64]
                 const float* __restrict__ b_emb,      // [64]
                 const float* __restrict__ b_enc,      // [512]
                 const float* __restrict__ b_dec,      // [512]
                 const float* __restrict__ b_out,      // [5]
                 const _Float16* __restrict__ wih_enc, // frags [2][32][32][16]
                 const _Float16* __restrict__ whh_enc, // frags [4][32][32][16]
                 const _Float16* __restrict__ wih_dec,
                 const _Float16* __restrict__ whh_dec,
                 const _Float16* __restrict__ wout_f,  // frags [4][32][16]
                 float* __restrict__ out)              // normals then pos
{
  __shared__ __attribute__((aligned(32))) _Float16 s_embA[4 * 2 * 32 * 16]; // 8KB
  __shared__ __attribute__((aligned(32))) _Float16 s_hA  [4 * 4 * 32 * 16]; // 16KB
  __shared__ float s_h32 [TM * HIDD];   // 32KB
  __shared__ float s_raw [TM * 5];      // head output (raw)
  __shared__ float s_vel [TM * 2];
  __shared__ float s_obs2[TM * 2];
  __shared__ float s_mask[TM];
  __shared__ float s_pos1[TM * 2];
  __shared__ float s_pos2[TM * 2];

  const int tid    = threadIdx.x;
  const int lane   = tid & 31;
  const int w      = tid >> 5;            // wave id == hid slab (0..7)
  const int wgbase = blockIdx.x * TM;

  // zero-init recurrent state
  for (int i = tid; i < 4 * 4 * 32 * 16; i += BLK) s_hA[i] = (_Float16)0.0f;
  for (int i = tid; i < TM * HIDD; i += BLK) s_h32[i] = 0.0f;
  float c_reg[4][8];
#pragma unroll
  for (int jj = 0; jj < 4; ++jj)
#pragma unroll
    for (int r = 0; r < 8; ++r) c_reg[jj][r] = 0.0f;
  __syncthreads();

  for (int t = 0; t < TOT_STEPS; ++t) {
    const bool enc = (t < ENC_STEPS);

    // ---- stage A: velocity / mask / obs2 (one thread per track) ----
    if (tid < TM) {
      int g = wgbase + tid;
      float o1x, o1y, o2x, o2y;
      if (enc) {
        if (g < NTRACK) {
          const float* p1 = observed + ((size_t)t * NTRACK + g) * 2;
          const float* p2 = observed + ((size_t)(t + 1) * NTRACK + g) * 2;
          o1x = p1[0]; o1y = p1[1]; o2x = p2[0]; o2y = p2[1];
        } else { o1x = o1y = o2x = o2y = 0.0f; }
      } else {
        o1x = s_pos2[tid * 2 + 0]; o1y = s_pos2[tid * 2 + 1];
        o2x = s_pos1[tid * 2 + 0]; o2y = s_pos1[tid * 2 + 1];
      }
      bool mk = (g < NTRACK) && !(o1x != o1x) && !(o2x != o2x);
      s_mask[tid]         = mk ? 1.0f : 0.0f;
      s_vel [tid * 2 + 0] = mk ? (o2x - o1x) : 0.0f;
      s_vel [tid * 2 + 1] = mk ? (o2y - o1y) : 0.0f;
      s_obs2[tid * 2 + 0] = o2x;
      s_obs2[tid * 2 + 1] = o2y;
    }

    // warm L2 with decoder weights just before the phase switch
    if (t == ENC_STEPS - 1) {
      const _Float16* pf = wih_dec;           // wih_dec..whh_dec contiguous
      for (int off = tid; off < (32768 + 65536) / 32; off += BLK)
        __builtin_prefetch(pf + (size_t)off * 32, 0, 1);
    }
    __syncthreads();

    // ---- stage B: embedding -> f16 A fragments ----
    for (int idx = tid; idx < TM * EMBD; idx += BLK) {
      int tr = idx >> 6, e = idx & 63;
      float v = 4.0f * s_vel[tr * 2 + 0] * W_emb[e] +
                4.0f * s_vel[tr * 2 + 1] * W_emb[64 + e] + b_emb[e];
      v = fmaxf(v, 0.0f);
      s_embA[a_frag_addr(tr >> 4, 2, tr & 15, e)] = (_Float16)v;
    }
    __syncthreads();

    const _Float16* wih  = enc ? wih_enc : wih_dec;
    const _Float16* whh  = enc ? whh_enc : whh_dec;
    const float*    bias = enc ? b_enc   : b_dec;

    // ---- WMMA stage: B frags loaded once per K-block, reused over 4 M-tiles
    v8f acc[4][4];
#pragma unroll
    for (int jj = 0; jj < 4; ++jj)
#pragma unroll
      for (int gi = 0; gi < 4; ++gi) acc[jj][gi] = 0.0f;

    // emb @ W_ih  (K = 64 -> 2 K-blocks)
#pragma unroll
    for (int kb = 0; kb < 2; ++kb) {
      v16h b[4];
#pragma unroll
      for (int gi = 0; gi < 4; ++gi)
        b[gi] = *(const v16h*)(wih +
                 (((kb * 32 + gi * 8 + w) * 32 + lane) << 4));
#pragma unroll
      for (int jj = 0; jj < 4; ++jj) {
        v16h a = *(const v16h*)&s_embA[(((jj * 2 + kb) * 32 + lane) << 4)];
#pragma unroll
        for (int gi = 0; gi < 4; ++gi)
          acc[jj][gi] = __builtin_amdgcn_wmma_f32_16x16x32_f16(
              false, a, false, b[gi], (short)0, acc[jj][gi], false, false);
      }
    }
    // h @ W_hh  (K = 128 -> 4 K-blocks)
#pragma unroll
    for (int kb = 0; kb < 4; ++kb) {
      v16h b[4];
#pragma unroll
      for (int gi = 0; gi < 4; ++gi)
        b[gi] = *(const v16h*)(whh +
                 (((kb * 32 + gi * 8 + w) * 32 + lane) << 4));
#pragma unroll
      for (int jj = 0; jj < 4; ++jj) {
        v16h a = *(const v16h*)&s_hA[(((jj * 4 + kb) * 32 + lane) << 4)];
#pragma unroll
        for (int gi = 0; gi < 4; ++gi)
          acc[jj][gi] = __builtin_amdgcn_wmma_f32_16x16x32_f16(
              false, a, false, b[gi], (short)0, acc[jj][gi], false, false);
      }
    }

    // ---- LSTM pointwise on accumulator layout:
    //   element (VGPR r, lane L) -> row m = r + 8*(L>>4), col = slab*16+(L&15)
    {
      const int col = w * 16 + (lane & 15);
      const float bi = bias[0 * 128 + col];
      const float bf = bias[1 * 128 + col];
      const float bg = bias[2 * 128 + col];
      const float bo = bias[3 * 128 + col];
#pragma unroll
      for (int jj = 0; jj < 4; ++jj) {
#pragma unroll
        for (int r = 0; r < 8; ++r) {
          int row  = jj * 16 + r + ((lane >> 4) << 3);
          float m  = s_mask[row];
          float iv = acc[jj][0][r] + bi;
          float fv = acc[jj][1][r] + bf;
          float gv = acc[jj][2][r] + bg;
          float ov = acc[jj][3][r] + bo;
          float cold = c_reg[jj][r];
          float cn   = sigm_fast(fv) * cold + sigm_fast(iv) * tanh_fast(gv);
          float hn   = sigm_fast(ov) * tanh_fast(cn);
          c_reg[jj][r] = (m > 0.5f) ? cn : cold;
          float hold = s_h32[row * HIDD + col];
          s_h32[row * HIDD + col] = (m > 0.5f) ? hn : hold;
        }
      }
    }
    __syncthreads();

    // ---- rebuild f16 A fragments of h for the next step / head ----
    for (int idx = tid; idx < TM * HIDD; idx += BLK) {
      int tr = idx >> 7, k = idx & 127;
      s_hA[a_frag_addr(tr >> 4, 4, tr & 15, k)] =
          (_Float16)s_h32[tr * HIDD + k];
    }
    __syncthreads();

    // ---- output head via WMMA: waves 0..3 each do one M-tile ----
    if (w < 4) {
      const int jj = w;
      v8f acco = 0.0f;
#pragma unroll
      for (int kb = 0; kb < 4; ++kb) {
        v16h a = *(const v16h*)&s_hA[(((jj * 4 + kb) * 32 + lane) << 4)];
        v16h b = *(const v16h*)(wout_f + (((kb * 32) + lane) << 4));
        acco = __builtin_amdgcn_wmma_f32_16x16x32_f16(
            false, a, false, b, (short)0, acco, false, false);
      }
      int j = lane & 15;
      if (j < 5) {
        float bj = b_out[j];
#pragma unroll
        for (int r = 0; r < 8; ++r) {
          int row = jj * 16 + r + ((lane >> 4) << 3);
          s_raw[row * 5 + j] = acco[r] + bj;
        }
      }
    }
    __syncthreads();

    // ---- tail: transform, store outputs, advance position carry ----
    if (tid < TM) {
      int g = wgbase + tid;
      if (g < NTRACK) {
        bool mk = s_mask[tid] > 0.5f;
        const float QNAN = __builtin_nanf("");
        float n0, n1, n2, n3, n4;
        if (mk) {
          n0 = s_raw[tid * 5 + 0];
          n1 = s_raw[tid * 5 + 1];
          n2 = 0.01f + 0.2f * softplus_fast(s_raw[tid * 5 + 2]);
          n3 = 0.01f + 0.2f * softplus_fast(s_raw[tid * 5 + 3]);
          n4 = 0.7f * sigm_fast(s_raw[tid * 5 + 4]);
        } else {
          n0 = n1 = n2 = n3 = n4 = QNAN;
        }
        float px = s_obs2[tid * 2 + 0] + n0;
        float py = s_obs2[tid * 2 + 1] + n1;

        size_t ob = (size_t)t * NTRACK * 5 + (size_t)g * 5;
        out[ob + 0] = n0; out[ob + 1] = n1; out[ob + 2] = n2;
        out[ob + 3] = n3; out[ob + 4] = n4;
        size_t pb = (size_t)TOT_STEPS * NTRACK * 5 +
                    (size_t)t * NTRACK * 2 + (size_t)g * 2;
        out[pb + 0] = px; out[pb + 1] = py;

        // shift position carry (p2 <- p1, p1 <- pos)
        s_pos2[tid * 2 + 0] = s_pos1[tid * 2 + 0];
        s_pos2[tid * 2 + 1] = s_pos1[tid * 2 + 1];
        s_pos1[tid * 2 + 0] = px;
        s_pos1[tid * 2 + 1] = py;
      }
    }
    __syncthreads();
  }
}

extern "C" void kernel_launch(void* const* d_in, const int* in_sizes, int n_in,
                              void* d_out, int out_size, void* d_ws, size_t ws_size,
                              hipStream_t stream) {
  (void)in_sizes; (void)n_in; (void)out_size; (void)ws_size;
  const float* observed = (const float*)d_in[0];
  // d_in[1]=goals, d_in[2]=batch_split, d_in[3]=n_predict : unused (constants)
  const float* W_emb    = (const float*)d_in[4];
  const float* b_emb    = (const float*)d_in[5];
  const float* W_ih_enc = (const float*)d_in[6];
  const float* W_hh_enc = (const float*)d_in[7];
  const float* b_enc    = (const float*)d_in[8];
  const float* W_ih_dec = (const float*)d_in[9];
  const float* W_hh_dec = (const float*)d_in[10];
  const float* b_dec    = (const float*)d_in[11];
  const float* W_out    = (const float*)d_in[12];
  const float* b_out    = (const float*)d_in[13];
  float* out = (float*)d_out;

  // Workspace layout (halfs): swizzled f16 weight fragments, ~388KB total.
  _Float16* ws = (_Float16*)d_ws;
  _Float16* wih_e = ws;                  // 2*32*32*16 = 32768
  _Float16* whh_e = wih_e + 32768;       // 4*32*32*16 = 65536
  _Float16* wih_d = whh_e + 65536;       // 32768
  _Float16* whh_d = wih_d + 32768;       // 65536
  _Float16* wout  = whh_d + 65536;       // 4*32*16    = 2048

  wprep_kernel<<<128, 256, 0, stream>>>(W_ih_enc, wih_e, EMBD);
  wprep_kernel<<<256, 256, 0, stream>>>(W_hh_enc, whh_e, HIDD);
  wprep_kernel<<<128, 256, 0, stream>>>(W_ih_dec, wih_d, EMBD);
  wprep_kernel<<<256, 256, 0, stream>>>(W_hh_dec, whh_d, HIDD);
  wprep_out_kernel<<<8, 256, 0, stream>>>(W_out, wout);

  int nblocks = (NTRACK + TM - 1) / TM;  // 313
  lstm_kernel<<<nblocks, BLK, 0, stream>>>(
      observed, W_emb, b_emb, b_enc, b_dec, b_out,
      wih_e, whh_e, wih_d, whh_d, wout, out);
}
